// Block_22497038696617
// MI455X (gfx1250) — matrix operand order repaired and checked by reading, not compiled, not run on hardware
//
#include <hip/hip_runtime.h>
#include <math.h>

// ---------------- shapes ----------------
#define BB 2
#define NN 2048
#define CC 768
#define HH 12
#define HD 64
#define HID 3072
#define BH (BB*HH)          // 24

typedef __attribute__((ext_vector_type(16))) _Float16 v16h;
typedef __attribute__((ext_vector_type(8)))  float    v8f;

union F16x16 { v16h v; float4 f[2]; };

__device__ __forceinline__ v8f wmma16(v16h a, v16h b, v8f c) {
  // D = A(16x32 f16) * B(32x16 f16) + C(16x16 f32)
  return __builtin_amdgcn_wmma_f32_16x16x32_f16(false, a, false, b, (short)0, c,
                                                false, false);
}

// A fragment (16x32, 16-bit): lane m%16 = row; half(lane>>4)=0 -> K 0..7,16..23 ; half=1 -> K 8..15,24..31
__device__ __forceinline__ v16h ldA16(const _Float16* A, int lda, int m, int kc, int hf) {
  const _Float16* p = A + (size_t)m * lda + kc + 8 * hf;
  F16x16 u;
  u.f[0] = *(const float4*)(p);
  u.f[1] = *(const float4*)(p + 16);
  return u.v;
}

// B fragment (32x16, 16-bit): lane n%16 = column; half=0 -> K 0..7,16..23 ; half=1 -> K 8..15,24..31
// B stored as [N, K] row-major (i.e. B^T of the math matrix), so a column is contiguous.
__device__ __forceinline__ v16h ldB16(const _Float16* B, int ldb, int n, int kc, int hf) {
  const _Float16* p = B + (size_t)n * ldb + kc + 8 * hf;
  F16x16 u;
  u.f[0] = *(const float4*)(p);
  u.f[1] = *(const float4*)(p + 16);
  return u.v;
}

// ---------- generic WMMA GEMM core: acc[MW][NW] += A[M,K]*B[N,K]^T ----------
template <int MW, int NW>
__device__ __forceinline__ void gemm_core(const _Float16* A, const _Float16* B, int K,
                                          int m0, int n0, int l16, int hf,
                                          v8f acc[MW][NW]) {
  for (int kc = 0; kc < K; kc += 32) {
    v16h af[MW], bf[NW];
#pragma unroll
    for (int mt = 0; mt < MW; ++mt) af[mt] = ldA16(A, K, m0 + 16 * mt + l16, kc, hf);
#pragma unroll
    for (int nt = 0; nt < NW; ++nt) bf[nt] = ldB16(B, K, n0 + 16 * nt + l16, kc, hf);
#pragma unroll
    for (int mt = 0; mt < MW; ++mt)
#pragma unroll
      for (int nt = 0; nt < NW; ++nt)
        acc[mt][nt] = wmma16(af[mt], bf[nt], acc[mt][nt]);
  }
}

#define GEMM_PROLOG(KDIM)                                      \
  const int lane = threadIdx.x & 31, wave = threadIdx.x >> 5;  \
  const int l16 = lane & 15, hf = lane >> 4;                   \
  const int m0 = blockIdx.y * 128 + (wave & 3) * 32;           \
  const int n0 = blockIdx.x * 128 + (wave >> 2) * 64;          \
  v8f acc[2][4];                                               \
  {                                                            \
    v8f z = {0, 0, 0, 0, 0, 0, 0, 0};                          \
    _Pragma("unroll") for (int i = 0; i < 2; ++i)              \
        _Pragma("unroll") for (int j = 0; j < 4; ++j)          \
            acc[i][j] = z;                                     \
  }                                                            \
  gemm_core<2, 4>(A, Bm, (KDIM), m0, n0, l16, hf, acc);

// C/D layout: VGPR r -> row m0+16*mt + r + 8*hf ; col n0+16*nt + l16

// ------------- QKV GEMM: h16[4096,768] x qkv_w^T -> q/k/v (f16, q scaled) -------------
__global__ __launch_bounds__(256) void k_gemm_qkv(const _Float16* A, const _Float16* Bm,
                                                  _Float16* q, _Float16* k, _Float16* v) {
  GEMM_PROLOG(CC)
#pragma unroll
  for (int mt = 0; mt < 2; ++mt)
#pragma unroll
    for (int nt = 0; nt < 4; ++nt)
#pragma unroll
      for (int r = 0; r < 8; ++r) {
        int m = m0 + 16 * mt + r + 8 * hf;
        int c = n0 + 16 * nt + l16;
        float val = acc[mt][nt][r];
        int which = c / CC;
        int hd = (c % CC) >> 6;
        int d  = c & 63;
        int b_ = m >> 11, nq = m & (NN - 1);
        size_t idx = ((size_t)(b_ * HH + hd) * NN + nq) * HD + d;
        if (which == 0)      q[idx] = (_Float16)(val * 0.125f);   // HD^-0.5
        else if (which == 1) k[idx] = (_Float16)val;
        else                 v[idx] = (_Float16)val;
      }
}

// ------------- proj GEMM + bias + residual -> x1 (f32) -------------
__global__ __launch_bounds__(256) void k_gemm_proj(const _Float16* A, const _Float16* Bm,
                                                   const float* xin, const float* bias,
                                                   float* x1) {
  GEMM_PROLOG(CC)
#pragma unroll
  for (int mt = 0; mt < 2; ++mt)
#pragma unroll
    for (int nt = 0; nt < 4; ++nt)
#pragma unroll
      for (int r = 0; r < 8; ++r) {
        int m = m0 + 16 * mt + r + 8 * hf;
        int c = n0 + 16 * nt + l16;
        size_t idx = (size_t)m * CC + c;
        x1[idx] = xin[idx] + acc[mt][nt][r] + bias[c];
      }
}

// ------------- FC1 GEMM + bias + exact GELU -> f16 -------------
__global__ __launch_bounds__(256) void k_gemm_fc1(const _Float16* A, const _Float16* Bm,
                                                  const float* bias, _Float16* out) {
  GEMM_PROLOG(CC)
#pragma unroll
  for (int mt = 0; mt < 2; ++mt)
#pragma unroll
    for (int nt = 0; nt < 4; ++nt)
#pragma unroll
      for (int r = 0; r < 8; ++r) {
        int m = m0 + 16 * mt + r + 8 * hf;
        int c = n0 + 16 * nt + l16;
        float t = acc[mt][nt][r] + bias[c];
        float g = 0.5f * t * (1.0f + erff(t * 0.70710678118654752f));
        out[(size_t)m * HID + c] = (_Float16)g;
      }
}

// ------------- FC2 GEMM + bias + residual -> f32 out -------------
__global__ __launch_bounds__(256) void k_gemm_fc2(const _Float16* A, const _Float16* Bm,
                                                  const float* bias, const float* x1,
                                                  float* out) {
  GEMM_PROLOG(HID)
#pragma unroll
  for (int mt = 0; mt < 2; ++mt)
#pragma unroll
    for (int nt = 0; nt < 4; ++nt)
#pragma unroll
      for (int r = 0; r < 8; ++r) {
        int m = m0 + 16 * mt + r + 8 * hf;
        int c = n0 + 16 * nt + l16;
        size_t idx = (size_t)m * CC + c;
        out[idx] = x1[idx] + acc[mt][nt][r] + bias[c];
      }
}

// ------------- LayerNorm (one row per block) -> f16 -------------
__global__ __launch_bounds__(256) void k_ln(const float* x, const float* g,
                                            const float* be, _Float16* out) {
  __shared__ float red[256];
  const int row = blockIdx.x, tid = threadIdx.x;
  const float* xr = x + (size_t)row * CC;
  float s = 0.f;
  for (int i = tid; i < CC; i += 256) s += xr[i];
  red[tid] = s; __syncthreads();
  for (int st = 128; st > 0; st >>= 1) { if (tid < st) red[tid] += red[tid + st]; __syncthreads(); }
  float mean = red[0] * (1.f / CC);
  __syncthreads();
  float vv = 0.f;
  for (int i = tid; i < CC; i += 256) { float d = xr[i] - mean; vv += d * d; }
  red[tid] = vv; __syncthreads();
  for (int st = 128; st > 0; st >>= 1) { if (tid < st) red[tid] += red[tid + st]; __syncthreads(); }
  float inv = rsqrtf(red[0] * (1.f / CC) + 1e-5f);
  for (int i = tid; i < CC; i += 256)
    out[(size_t)row * CC + i] = (_Float16)((xr[i] - mean) * inv * g[i] + be[i]);
}

// ------------- f32 -> f16 cast -------------
__global__ void k_cast(const float* s, _Float16* d, int n) {
  int i = blockIdx.x * 256 + threadIdx.x;
  if (i < n) d[i] = (_Float16)s[i];
}

// ------------- Vsum[b,h,d] = sum_n V  (for the eps/n correction) -------------
__global__ __launch_bounds__(256) void k_vsum(const _Float16* v, float* vs) {
  int t = blockIdx.x * 256 + threadIdx.x;
  if (t >= BH * HD) return;
  int bh = t >> 6, d = t & 63;
  const _Float16* p = v + (size_t)bh * NN * HD + d;
  float s = 0.f;
  for (int n = 0; n < NN; ++n) s += (float)p[(size_t)n * HD];
  vs[t] = s;
}

// ------------- flash attention with policy mask -------------
// grid: (BH, NN/64), block 128 (4 waves, each owns 16 query rows)
__global__ __launch_bounds__(128) void k_attn(const _Float16* q16, const _Float16* k16,
                                              const _Float16* v16, const float* policy,
                                              const float* vsum, _Float16* o16) {
  __shared__ _Float16 VT[64 * 32];       // [d][key]  V chunk transposed
  __shared__ _Float16 Pst[4 * 16 * 32];  // per-wave P staging [q][key]
  const int lane = threadIdx.x & 31, wave = threadIdx.x >> 5;
  const int l16 = lane & 15, hf = lane >> 4;
  const int bh = blockIdx.x;
  const int b = bh / HH, h = bh % HH;
  const int qbase = blockIdx.y * 64 + wave * 16;

  // Q A-fragments (K = 64 -> two 16x32 chunks), loaded once
  const _Float16* qrow = q16 + ((size_t)bh * NN + qbase + l16) * HD;
  v16h aQ[2];
#pragma unroll
  for (int j = 0; j < 2; ++j) {
    F16x16 u;
    u.f[0] = *(const float4*)(qrow + 32 * j + 8 * hf);
    u.f[1] = *(const float4*)(qrow + 32 * j + 16 + 8 * hf);
    aQ[j] = u.v;
  }
  v8f z = {0, 0, 0, 0, 0, 0, 0, 0};
  v8f oacc[4] = {z, z, z, z};
  float rowm[8], rowd[8];
#pragma unroll
  for (int r = 0; r < 8; ++r) { rowm[r] = -1e30f; rowd[r] = 0.f; }
  const float* pol = policy + b * NN;
  _Float16* pw = Pst + wave * (16 * 32);

  for (int kb = 0; kb < NN; kb += 32) {
    __syncthreads();  // protect VT reuse across iterations
    // cooperative transpose-load of V chunk: VT[d][key]
    for (int i = threadIdx.x; i < 32 * 64; i += 128) {
      int key = i >> 6, d = i & 63;
      VT[d * 32 + key] = v16[((size_t)bh * NN + kb + key) * HD + d];
    }
    __syncthreads();

    // S = Q * K^T for two 16-key subtiles (K fragments straight from global: rows contiguous)
    v8f S[2];
#pragma unroll
    for (int s = 0; s < 2; ++s) {
      const _Float16* krow = k16 + ((size_t)bh * NN + kb + s * 16 + l16) * HD;
      v8f acc = z;
#pragma unroll
      for (int j = 0; j < 2; ++j) {
        F16x16 u;
        const _Float16* p = krow + 32 * j + 8 * hf;
        u.f[0] = *(const float4*)p;
        u.f[1] = *(const float4*)(p + 16);
        acc = wmma16(aQ[j], u.v, acc);
      }
      S[s] = acc;
    }

    // online softmax with policy gating (self always attends)
    float p0 = pol[kb + l16];
    float p1 = pol[kb + 16 + l16];
#pragma unroll
    for (int r = 0; r < 8; ++r) {
      float s0 = S[0][r], s1 = S[1][r];
      float cm = fmaxf(s0, s1);
#pragma unroll
      for (int msk = 1; msk <= 8; msk <<= 1) cm = fmaxf(cm, __shfl_xor(cm, msk, 32));
      float mnew = fmaxf(rowm[r], cm);
      float sc = expf(rowm[r] - mnew);
      int qi = qbase + r + 8 * hf;
      float ap0 = (kb + l16 == qi) ? 1.f : p0;
      float ap1 = (kb + 16 + l16 == qi) ? 1.f : p1;
      float e0 = expf(s0 - mnew) * ap0;
      float e1 = expf(s1 - mnew) * ap1;
      float es = e0 + e1;
#pragma unroll
      for (int msk = 1; msk <= 8; msk <<= 1) es += __shfl_xor(es, msk, 32);
      rowd[r] = rowd[r] * sc + es;
      rowm[r] = mnew;
#pragma unroll
      for (int t = 0; t < 4; ++t) oacc[t][r] *= sc;  // rescale running numerator
      pw[(r + 8 * hf) * 32 + l16]      = (_Float16)e0;  // stage P (C-layout -> LDS)
      pw[(r + 8 * hf) * 32 + 16 + l16] = (_Float16)e1;
    }
    asm volatile("s_wait_dscnt 0" ::: "memory");  // cross-lane LDS RAW within wave

    // A-fragment of P from LDS; O += P * V
    F16x16 up;
    const _Float16* pr = pw + l16 * 32;
    up.f[0] = *(const float4*)(pr + 8 * hf);
    up.f[1] = *(const float4*)(pr + 16 + 8 * hf);
#pragma unroll
    for (int t = 0; t < 4; ++t) {
      F16x16 ub;
      const _Float16* vp = VT + (16 * t + l16) * 32 + 8 * hf;
      ub.f[0] = *(const float4*)vp;
      ub.f[1] = *(const float4*)(vp + 16);
      oacc[t] = wmma16(up.v, ub.v, oacc[t]);
    }
  }

  // O = (num + (eps/n)*Vsum) / (den + eps), write f16 [b, n, h*64+d]
  const float epsn = 1e-6f / (float)NN;
#pragma unroll
  for (int t = 0; t < 4; ++t)
#pragma unroll
    for (int r = 0; r < 8; ++r) {
      int d = 16 * t + l16;
      int nq = qbase + r + 8 * hf;
      float val = (oacc[t][r] + epsn * vsum[bh * HD + d]) / (rowd[r] + 1e-6f);
      o16[((size_t)(b * NN + nq)) * CC + h * HD + d] = (_Float16)val;
    }
}

// ---------------- launch ----------------
extern "C" void kernel_launch(void* const* d_in, const int* in_sizes, int n_in,
                              void* d_out, int out_size, void* d_ws, size_t ws_size,
                              hipStream_t stream) {
  const float* x      = (const float*)d_in[0];
  const float* policy = (const float*)d_in[1];
  const float* ln1_g  = (const float*)d_in[2];
  const float* ln1_b  = (const float*)d_in[3];
  const float* qkv_w  = (const float*)d_in[4];
  const float* proj_w = (const float*)d_in[5];
  const float* proj_b = (const float*)d_in[6];
  const float* ln2_g  = (const float*)d_in[7];
  const float* ln2_b  = (const float*)d_in[8];
  const float* fc1_w  = (const float*)d_in[9];
  const float* fc1_b  = (const float*)d_in[10];
  const float* fc2_w  = (const float*)d_in[11];
  const float* fc2_b  = (const float*)d_in[12];
  float* out = (float*)d_out;

  char* ws = (char*)d_ws;
  size_t off = 0;
  auto alloc = [&](size_t bytes) -> void* {
    void* p = ws + off;
    off += (bytes + 255) & ~(size_t)255;
    return p;
  };
  const int M = BB * NN;  // 4096
  _Float16* qkv16 = (_Float16*)alloc((size_t)3 * CC * CC * 2);
  _Float16* proj16 = (_Float16*)alloc((size_t)CC * CC * 2);
  _Float16* fc116 = (_Float16*)alloc((size_t)HID * CC * 2);
  _Float16* fc216 = (_Float16*)alloc((size_t)CC * HID * 2);
  _Float16* h16 = (_Float16*)alloc((size_t)M * CC * 2);
  _Float16* q16 = (_Float16*)alloc((size_t)BH * NN * HD * 2);
  _Float16* k16 = (_Float16*)alloc((size_t)BH * NN * HD * 2);
  _Float16* v16 = (_Float16*)alloc((size_t)BH * NN * HD * 2);
  float*    vsumB = (float*)alloc((size_t)BH * HD * 4);
  _Float16* o16 = (_Float16*)alloc((size_t)M * CC * 2);
  float*    x1  = (float*)alloc((size_t)M * CC * 4);
  _Float16* h216 = (_Float16*)alloc((size_t)M * CC * 2);
  _Float16* h316 = (_Float16*)alloc((size_t)M * HID * 2);

  // weight casts f32 -> f16
  {
    int n;
    n = 3 * CC * CC; k_cast<<<(n + 255) / 256, 256, 0, stream>>>(qkv_w, qkv16, n);
    n = CC * CC;     k_cast<<<(n + 255) / 256, 256, 0, stream>>>(proj_w, proj16, n);
    n = HID * CC;    k_cast<<<(n + 255) / 256, 256, 0, stream>>>(fc1_w, fc116, n);
    n = CC * HID;    k_cast<<<(n + 255) / 256, 256, 0, stream>>>(fc2_w, fc216, n);
  }

  // LN1 -> h16
  k_ln<<<M, 256, 0, stream>>>(x, ln1_g, ln1_b, h16);
  // QKV GEMM
  k_gemm_qkv<<<dim3((3 * CC) / 128, M / 128), 256, 0, stream>>>(h16, qkv16, q16, k16, v16);
  // V column sums
  k_vsum<<<(BH * HD + 255) / 256, 256, 0, stream>>>(v16, vsumB);
  // flash attention
  k_attn<<<dim3(BH, NN / 64), 128, 0, stream>>>(q16, k16, v16, policy, vsumB, o16);
  // proj + residual -> x1
  k_gemm_proj<<<dim3(CC / 128, M / 128), 256, 0, stream>>>(o16, proj16, x, proj_b, x1);
  // LN2 -> h216
  k_ln<<<M, 256, 0, stream>>>(x1, ln2_g, ln2_b, h216);
  // FC1 + GELU
  k_gemm_fc1<<<dim3(HID / 128, M / 128), 256, 0, stream>>>(h216, fc116, fc1_b, h316);
  // FC2 + residual -> out
  k_gemm_fc2<<<dim3(CC / 128, M / 128), 256, 0, stream>>>(h316, fc216, fc2_b, x1, out);
}
